// SelfAttnRNN_38972533244499
// MI455X (gfx1250) — matrix operand order repaired and checked by reading, not compiled, not run on hardware
//
#include <hip/hip_runtime.h>

typedef __attribute__((ext_vector_type(16))) __bf16 v16bf;
typedef __attribute__((ext_vector_type(8)))  float  v8f;

union BF16x16 { unsigned u[8]; v16bf v; };

#if defined(__has_builtin)
#  if __has_builtin(__builtin_amdgcn_cvt_pk_bf16_f32)
#    define HAVE_CVT_PK_BF16 1
#  else
#    define HAVE_CVT_PK_BF16 0
#  endif
#  if __has_builtin(__builtin_amdgcn_tanhf)
#    define HAVE_TANHF 1
#  else
#    define HAVE_TANHF 0
#  endif
#else
#  define HAVE_CVT_PK_BF16 0
#  define HAVE_TANHF 0
#endif

// pack two f32 -> packed bf16x2 (lo in [15:0], hi in [31:16])
__device__ __forceinline__ unsigned pack2(float lo, float hi) {
#if HAVE_CVT_PK_BF16
    auto p = __builtin_amdgcn_cvt_pk_bf16_f32(lo, hi);   // v_cvt_pk_bf16_f32
    return __builtin_bit_cast(unsigned, p);
#else
    // round-half-up to bf16, then pack the two high halves with one v_perm_b32
    unsigned a = __float_as_uint(lo) + 0x8000u;
    unsigned b = __float_as_uint(hi) + 0x8000u;
    return __builtin_amdgcn_perm(b, a, 0x07060302u);     // {b[31:16], a[31:16]}
#endif
}

__device__ __forceinline__ float fast_tanh(float x) {
#if HAVE_TANHF
    return __builtin_amdgcn_tanhf(x);                    // v_tanh_f32 (gfx1250)
#else
    float e = __expf(2.0f * x);
    float r = __builtin_amdgcn_rcpf(e + 1.0f);
    return fmaf(-2.0f, r, 1.0f);                          // 1 - 2/(e^{2x}+1)
#endif
}

// lane <-> lane^16 full swap (xor=0x10, and=0x1f, or=0)
#define SWAPX16(x) __builtin_amdgcn_ds_swizzle((x), 0x401F)

// Problem constants from the reference: b=256, w=32, m=512, hid=64.
#define SEQ_LEN 32
#define MDIM    512

__global__ __launch_bounds__(256) void selfattn_rnn_38972533244499(
    const float* __restrict__ x,     // (256, 32, 512)
    const float* __restrict__ W_ih,  // (64,)
    const float* __restrict__ b_ih,  // (64,)
    const float* __restrict__ W_hh,  // (64, 64) row-major
    const float* __restrict__ b_hh,  // (64,)
    const float* __restrict__ Wout,  // (64,)
    const float* __restrict__ bout,  // (1,)
    float* __restrict__ out,         // (131072,)  n = b_idx*512 + m_idx
    int n_tiles)
{
    const int lane  = threadIdx.x & 31;
    const int wave  = threadIdx.x >> 5;
    const int tile  = blockIdx.x * 8 + wave;     // 16 sequences per tile
    if (tile >= n_tiles) return;
    const int  l15    = lane & 15;
    const bool lohalf = lane < 16;
    const int  half8  = lohalf ? 0 : 8;

    // ---- W_hh -> 8 resident WMMA A-operands (bf16, A layout) ----
    // A[2t+k] covers rows j = 16t..16t+15, cols kdim = 32k..32k+31.
// A, this code's k order: lanes l and l+16 share row M=l&15; lane<16 holds K{+0..15},
// lane>=16 holds K{+16..31}; packed pairs, low element first.
    v16bf A[8];
    #pragma unroll
    for (int t = 0; t < 4; ++t) {
        const int row = 16 * t + l15;
        #pragma unroll
        for (int k = 0; k < 2; ++k) {
            const float* p = W_hh + row * 64 + 32 * k + 2 * half8;
            BF16x16 a;
            #pragma unroll
            for (int j = 0; j < 4; ++j) a.u[j]     = pack2(p[2*j],      p[2*j+1]);
            #pragma unroll
            for (int j = 0; j < 4; ++j) a.u[4 + j] = pack2(p[8 + 2*j], p[8 + 2*j+1]);
            A[2 * t + k] = a.v;
        }
    }

    // ---- per-lane input weight / bias in D layout:
    // D tile t, vgpr r on this lane holds hidden index j = 16t + half8 + r
    float wih[4][8], bsum[4][8];
    #pragma unroll
    for (int t = 0; t < 4; ++t)
        #pragma unroll
        for (int r = 0; r < 8; ++r) {
            const int j = 16 * t + half8 + r;
            wih[t][r]  = W_ih[j];
            bsum[t][r] = b_ih[j] + b_hh[j];
        }

    // ---- running hidden-state sum, f32, D layout; 4 stacked 16x16 tiles ----
    v8f hsum[4], hx[4];
    #pragma unroll
    for (int t = 0; t < 4; ++t)
        #pragma unroll
        for (int r = 0; r < 8; ++r) { hsum[t][r] = 0.0f; hx[t][r] = 0.0f; }

    // x addressing: sequence n = tile*16 + l15; x[b_idx, step, m_idx]
    const int n     = tile * 16 + l15;
    const int b_idx = n / MDIM;
    const int m_idx = n % MDIM;
    const float* xp = x + (size_t)b_idx * SEQ_LEN * MDIM + m_idx;

    #pragma unroll 1
    for (int step = 0; step < SEQ_LEN; ++step) {
        const float xt = xp[step * MDIM];

        // ---- D-layout f32 hsum -> B-layout bf16 operands (register-only) ----
        // B (32x16), k order as in A: lanes 0-15 hold col N=l, K=0..15; lanes 16-31 hold N=l-16,
        // K=16..31; packed pairs. Cross-half rows come from lane^16: select the
        // OPPOSITE tile's pack first, swizzle once, then blend per half.
        unsigned q[4][4];
        #pragma unroll
        for (int t = 0; t < 4; ++t)
            #pragma unroll
            for (int j = 0; j < 4; ++j)
                q[t][j] = pack2(hsum[t][2*j], hsum[t][2*j + 1]);

        BF16x16 B0, B1;
        #pragma unroll
        for (int j = 0; j < 4; ++j) {
            unsigned e0 = lohalf ? q[1][j] : q[0][j];
            unsigned sw0 = (unsigned)SWAPX16((int)e0);   // lo: q0@l+16, hi: q1@l-16
            B0.u[j]     = lohalf ? q[0][j] : sw0;        // K 0..7   | 16..23
            B0.u[4 + j] = lohalf ? sw0     : q[1][j];    // K 8..15  | 24..31
            unsigned e1 = lohalf ? q[3][j] : q[2][j];
            unsigned sw1 = (unsigned)SWAPX16((int)e1);
            B1.u[j]     = lohalf ? q[2][j] : sw1;        // K 32..39 | 48..55
            B1.u[4 + j] = lohalf ? sw1     : q[3][j];    // K 40..47 | 56..63
        }

        // ---- hx^T = tanh( W_hh x hsum^T + x_t*W_ih + bias ), per 16-row tile ----
        #pragma unroll
        for (int t = 0; t < 4; ++t) {
            v8f acc;
            #pragma unroll
            for (int r = 0; r < 8; ++r) acc[r] = fmaf(xt, wih[t][r], bsum[t][r]);
            acc = __builtin_amdgcn_wmma_f32_16x16x32_bf16(
                      false, A[2*t + 0], false, B0.v, (short)0, acc, false, false);
            acc = __builtin_amdgcn_wmma_f32_16x16x32_bf16(
                      false, A[2*t + 1], false, B1.v, (short)0, acc, false, false);
            #pragma unroll
            for (int r = 0; r < 8; ++r) acc[r] = fast_tanh(acc[r]);
            hx[t] = acc;
            #pragma unroll
            for (int r = 0; r < 8; ++r) hsum[t][r] += acc[r];
        }
    }

    // ---- out[n] = Wout . hx_final[:, n] + bout ----
    float partial = 0.0f;
    #pragma unroll
    for (int t = 0; t < 4; ++t) {
        const float* wp = Wout + 16 * t + half8;
        #pragma unroll
        for (int r = 0; r < 8; ++r) partial = fmaf(wp[r], hx[t][r], partial);
    }
    partial += __int_as_float(SWAPX16(__float_as_int(partial)));
    if (lohalf) out[tile * 16 + l15] = partial + bout[0];
}

extern "C" void kernel_launch(void* const* d_in, const int* in_sizes, int n_in,
                              void* d_out, int out_size, void* d_ws, size_t ws_size,
                              hipStream_t stream) {
    (void)n_in; (void)d_ws; (void)ws_size; (void)in_sizes;
    const float* x    = (const float*)d_in[0];
    const float* W_ih = (const float*)d_in[1];
    const float* b_ih = (const float*)d_in[2];
    const float* W_hh = (const float*)d_in[3];
    const float* b_hh = (const float*)d_in[4];
    // d_in[5..8] = V, Wx, Wtlt, Wh: provably unused (softmax over size-1 axis == 1)
    const float* Wout = (const float*)d_in[9];
    const float* bout = (const float*)d_in[10];
    float* out = (float*)d_out;

    const int n_seq  = out_size;          // 256*512 = 131072
    const int tiles  = (n_seq + 15) / 16; // 8192 waves of 16 sequences
    const int blocks = (tiles + 7) / 8;   // 8 waves / 256-thread block

    selfattn_rnn_38972533244499<<<blocks, 256, 0, stream>>>(
        x, W_ih, b_ih, W_hh, b_hh, Wout, bout, out, tiles);
}
